// _ATTR_NETWORK_18777597018548
// MI455X (gfx1250) — matrix operand order repaired and checked by reading, not compiled, not run on hardware
//
#include <hip/hip_runtime.h>
#include <hip/hip_bf16.h>
#include <stdint.h>

// ---- problem constants (match reference) ----
#define B_   256
#define S_   128
#define E_   256
#define H_   4
#define HD_  64
#define L_   2
#define F_   1024
#define P_   100
#define NN_  500
#define T_   600      // P_+NN_
#define O3E_ 768      // 3*E

typedef __attribute__((ext_vector_type(16))) __bf16 v16bf;
typedef __attribute__((ext_vector_type(8)))  float  v8f;

__device__ __forceinline__ unsigned short f2bf(float f) {
  unsigned u = __float_as_uint(f);
  u += 0x7FFFu + ((u >> 16) & 1u);     // round-to-nearest-even
  return (unsigned short)(u >> 16);
}

__device__ __forceinline__ unsigned pack2bf(float a, float b) {
  return (unsigned)f2bf(a) | ((unsigned)f2bf(b) << 16);
}

union FragU { v16bf v; uint4 q[2]; };

// A-fragment (16x32 bf16, ISA 7.12.2): lanes 0-15 K in {0..7,16..23}; lanes 16-31 {8..15,24..31}.
__device__ __forceinline__ v16bf frag_a(const unsigned short* base, int ld, int lane) {
  const int r  = lane & 15;
  const int kb = (lane >> 4) << 3;
  FragU f;
  f.q[0] = *(const uint4*)(base + r * ld + kb);
  f.q[1] = *(const uint4*)(base + r * ld + kb + 16);
  return f.v;
}

// B-fragment (32x16 bf16): column n=lane&15; lanes 0-15 K in {0..7,16..23}; lanes 16-31 {8..15,24..31}.
__device__ __forceinline__ v16bf frag_b(const unsigned short* base, int ld, int lane) {
  const int n  = lane & 15;
  const int kb = (lane >> 4) << 3;
  FragU f;
  f.q[0] = *(const uint4*)(base + n * ld + kb);
  f.q[1] = *(const uint4*)(base + n * ld + kb + 16);
  return f.v;
}

__device__ __forceinline__ v8f wmma_bf16(v16bf a, v16bf b, v8f c) {
  return __builtin_amdgcn_wmma_f32_16x16x32_bf16(false, a, false, b, (short)0, c, false, false);
}

// ---------------- generic C = A @ Bt^T + bias (optional ReLU) ----------------
// A:[M,K] f32 row-major, Bt:[N,K] f32 row-major. Block tile 128x64, 8 waves,
// each wave computes a 32x32 tile. Double-buffered LDS panels: global b128
// loads for panel k+1 are issued before the WMMA chain on panel k.
template<bool RELU>
__global__ __launch_bounds__(256)
void gemm_bt(const float* __restrict__ A, const float* __restrict__ Bt,
             const float* __restrict__ bias, float* __restrict__ C,
             int M, int N, int K) {
  __shared__ __align__(16) unsigned short As[2][128 * 32];
  __shared__ __align__(16) unsigned short Bs[2][64 * 32];
  const int t    = threadIdx.x;
  const int lane = t & 31;
  const int w    = t >> 5;
  const int bm   = blockIdx.x * 128;
  const int bn   = blockIdx.y * 64;
  const int m0   = (w & 3) * 32;
  const int n0   = (w >> 2) * 32;
  const int ra_r  = t >> 3,           ra_c = (t & 7) << 2;   // A: rows t/8, t/8+32.., col group
  const int rb_r  = (t & 511) >> 3;                          // B rows (two 256-chunks -> 0..63)

  float4 ra[4], rb[2];
  // prologue: panel 0 global -> regs -> LDS[0]
#pragma unroll
  for (int i = 0; i < 4; ++i)
    ra[i] = *(const float4*)(A + (size_t)(bm + ra_r + 32 * i) * K + ra_c);
#pragma unroll
  for (int i = 0; i < 2; ++i)
    rb[i] = *(const float4*)(Bt + (size_t)(bn + rb_r + 32 * i) * K + ra_c);
#pragma unroll
  for (int i = 0; i < 4; ++i) {
    uint2 u; u.x = pack2bf(ra[i].x, ra[i].y); u.y = pack2bf(ra[i].z, ra[i].w);
    *(uint2*)&As[0][(ra_r + 32 * i) * 32 + ra_c] = u;
  }
#pragma unroll
  for (int i = 0; i < 2; ++i) {
    uint2 u; u.x = pack2bf(rb[i].x, rb[i].y); u.y = pack2bf(rb[i].z, rb[i].w);
    *(uint2*)&Bs[0][(rb_r + 32 * i) * 32 + ra_c] = u;
  }
  __syncthreads();

  v8f c00 = {}, c01 = {}, c10 = {}, c11 = {};
  const int NK = K >> 5;
  for (int kk = 0; kk < NK; ++kk) {
    const int cur = kk & 1;
    const bool more = (kk + 1 < NK);
    if (more) {                        // issue next panel's global loads early
      const int k0 = (kk + 1) << 5;
#pragma unroll
      for (int i = 0; i < 4; ++i)
        ra[i] = *(const float4*)(A + (size_t)(bm + ra_r + 32 * i) * K + k0 + ra_c);
#pragma unroll
      for (int i = 0; i < 2; ++i)
        rb[i] = *(const float4*)(Bt + (size_t)(bn + rb_r + 32 * i) * K + k0 + ra_c);
    }
    v16bf a0 = frag_a(&As[cur][(m0 +  0) * 32], 32, lane);
    v16bf a1 = frag_a(&As[cur][(m0 + 16) * 32], 32, lane);
    v16bf b0 = frag_b(&Bs[cur][(n0 +  0) * 32], 32, lane);
    v16bf b1 = frag_b(&Bs[cur][(n0 + 16) * 32], 32, lane);
    c00 = wmma_bf16(a0, b0, c00);
    c01 = wmma_bf16(a0, b1, c01);
    c10 = wmma_bf16(a1, b0, c10);
    c11 = wmma_bf16(a1, b1, c11);
    if (more) {
      const int nxt = cur ^ 1;
#pragma unroll
      for (int i = 0; i < 4; ++i) {
        uint2 u; u.x = pack2bf(ra[i].x, ra[i].y); u.y = pack2bf(ra[i].z, ra[i].w);
        *(uint2*)&As[nxt][(ra_r + 32 * i) * 32 + ra_c] = u;
      }
#pragma unroll
      for (int i = 0; i < 2; ++i) {
        uint2 u; u.x = pack2bf(rb[i].x, rb[i].y); u.y = pack2bf(rb[i].z, rb[i].w);
        *(uint2*)&Bs[nxt][(rb_r + 32 * i) * 32 + ra_c] = u;
      }
      __syncthreads();
    }
  }

  const int cn = lane & 15;
  const int g8 = (lane >> 4) << 3;
#pragma unroll
  for (int tn = 0; tn < 2; ++tn) {
    int col = bn + n0 + tn * 16 + cn;
    float bs = bias[col];
    v8f cA = tn ? c01 : c00;   // m-tile 0
    v8f cB = tn ? c11 : c10;   // m-tile 1
#pragma unroll
    for (int j = 0; j < 8; ++j) {
      float vA = cA[j] + bs;
      float vB = cB[j] + bs;
      if (RELU) { vA = vA > 0.f ? vA : 0.f; vB = vB > 0.f ? vB : 0.f; }
      C[(size_t)(bm + m0 +      j + g8) * N + col] = vA;
      C[(size_t)(bm + m0 + 16 + j + g8) * N + col] = vB;
    }
  }
}

// ---------------- fused attention per (batch, head) ----------------
// qkv:[B,S,3E]; out:[B,S,E] (head h writes cols h*64..h*64+63)
__global__ __launch_bounds__(256)
void attention_kernel(const float* __restrict__ qkv, const int* __restrict__ lens,
                      float* __restrict__ out) {
  __shared__ __align__(16) unsigned short smem[24576];  // 48 KB
  unsigned short* Qs = smem;                      // [128][64]
  unsigned short* Ks = smem + 128 * 64;           // [128][64]
  unsigned short* Vt = smem + 2 * 128 * 64;       // [64][128]  (V transposed)
  unsigned short* Pb = smem;                      // [128][128] aliases Qs+Ks

  const int t = threadIdx.x, lane = t & 31, w = t >> 5;
  const int b = blockIdx.x / H_, h = blockIdx.x % H_;
  const float* base = qkv + (size_t)b * S_ * 3 * E_ + h * HD_;

  // stage Q, K (row-major bf16, packed b64 stores)
#pragma unroll
  for (int i = 0; i < 8; ++i) {
    int idx = t + 256 * i;                 // 0..2047
    int s = idx >> 4, c4 = (idx & 15) << 2;
    float4 q4 = *(const float4*)(base + (size_t)s * (3 * E_) + c4);
    float4 k4 = *(const float4*)(base + (size_t)s * (3 * E_) + E_ + c4);
    uint2 uq; uq.x = pack2bf(q4.x, q4.y); uq.y = pack2bf(q4.z, q4.w);
    *(uint2*)&Qs[s * 64 + c4] = uq;
    uint2 uk; uk.x = pack2bf(k4.x, k4.y); uk.y = pack2bf(k4.z, k4.w);
    *(uint2*)&Ks[s * 64 + c4] = uk;
  }
  // stage V transposed: Vt[n][s] = V[s][n]
#pragma unroll
  for (int i = 0; i < 32; ++i) {
    int idx = t + 256 * i;                 // 0..8191
    int s = idx >> 6, n = idx & 63;
    Vt[n * 128 + s] = f2bf(base[(size_t)s * (3 * E_) + 2 * E_ + n]);
  }
  __syncthreads();

  // scores: rows 16w..16w+15, 8 n-tiles, K=64 in 2 steps; B-frags preloaded 4-wide
  v8f sc[8];
#pragma unroll
  for (int nt = 0; nt < 8; ++nt) { v8f z = {}; sc[nt] = z; }
#pragma unroll
  for (int k0 = 0; k0 < 64; k0 += 32) {
    v16bf a = frag_a(&Qs[(w * 16) * 64 + k0], 64, lane);
#pragma unroll
    for (int g = 0; g < 2; ++g) {
      v16bf b0 = frag_b(&Ks[((g * 4 + 0) * 16) * 64 + k0], 64, lane);
      v16bf b1 = frag_b(&Ks[((g * 4 + 1) * 16) * 64 + k0], 64, lane);
      v16bf b2 = frag_b(&Ks[((g * 4 + 2) * 16) * 64 + k0], 64, lane);
      v16bf b3 = frag_b(&Ks[((g * 4 + 3) * 16) * 64 + k0], 64, lane);
      sc[g * 4 + 0] = wmma_bf16(a, b0, sc[g * 4 + 0]);
      sc[g * 4 + 1] = wmma_bf16(a, b1, sc[g * 4 + 1]);
      sc[g * 4 + 2] = wmma_bf16(a, b2, sc[g * 4 + 2]);
      sc[g * 4 + 3] = wmma_bf16(a, b3, sc[g * 4 + 3]);
    }
  }
  __syncthreads();   // done reading Qs/Ks; Pb may overwrite them after softmax

  // scale + key-padding mask + row softmax (row lives in one 16-lane half-group)
  const int len = lens[b];
  const int cn = lane & 15, g8 = (lane >> 4) << 3;
  float rmax[8], rsum[8];
#pragma unroll
  for (int j = 0; j < 8; ++j) { rmax[j] = -1e30f; rsum[j] = 0.f; }
#pragma unroll
  for (int nt = 0; nt < 8; ++nt) {
    float msk = ((nt * 16 + cn) < len) ? 0.f : -1e9f;
#pragma unroll
    for (int j = 0; j < 8; ++j) {
      float v = sc[nt][j] * 0.125f + msk;
      sc[nt][j] = v;
      rmax[j] = fmaxf(rmax[j], v);
    }
  }
#pragma unroll
  for (int off = 1; off < 16; off <<= 1)
#pragma unroll
    for (int j = 0; j < 8; ++j) rmax[j] = fmaxf(rmax[j], __shfl_xor(rmax[j], off, 32));
#pragma unroll
  for (int nt = 0; nt < 8; ++nt)
#pragma unroll
    for (int j = 0; j < 8; ++j) {
      float e = __expf(sc[nt][j] - rmax[j]);
      sc[nt][j] = e; rsum[j] += e;
    }
#pragma unroll
  for (int off = 1; off < 16; off <<= 1)
#pragma unroll
    for (int j = 0; j < 8; ++j) rsum[j] += __shfl_xor(rsum[j], off, 32);
#pragma unroll
  for (int j = 0; j < 8; ++j) rsum[j] = 1.f / rsum[j];

  // store P (bf16) into Pb
#pragma unroll
  for (int nt = 0; nt < 8; ++nt)
#pragma unroll
    for (int j = 0; j < 8; ++j)
      Pb[(w * 16 + j + g8) * 128 + nt * 16 + cn] = f2bf(sc[nt][j] * rsum[j]);
  __syncthreads();

  // O = P(128x128) @ V(128x64): rows 16w..16w+15, 4 n-tiles, 4 K steps; preload B-frags
  v8f o[4];
#pragma unroll
  for (int nt = 0; nt < 4; ++nt) { v8f z = {}; o[nt] = z; }
#pragma unroll
  for (int k0 = 0; k0 < 128; k0 += 32) {
    v16bf a  = frag_a(&Pb[(w * 16) * 128 + k0], 128, lane);
    v16bf b0 = frag_b(&Vt[( 0) * 128 + k0], 128, lane);
    v16bf b1 = frag_b(&Vt[(16) * 128 + k0], 128, lane);
    v16bf b2 = frag_b(&Vt[(32) * 128 + k0], 128, lane);
    v16bf b3 = frag_b(&Vt[(48) * 128 + k0], 128, lane);
    o[0] = wmma_bf16(a, b0, o[0]);
    o[1] = wmma_bf16(a, b1, o[1]);
    o[2] = wmma_bf16(a, b2, o[2]);
    o[3] = wmma_bf16(a, b3, o[3]);
  }
  float* obase = out + (size_t)b * S_ * E_ + h * HD_;
#pragma unroll
  for (int nt = 0; nt < 4; ++nt)
#pragma unroll
    for (int j = 0; j < 8; ++j)
      obase[(size_t)(w * 16 + j + g8) * E_ + nt * 16 + cn] = o[nt][j];
}

// ---------------- residual add + LayerNorm (one wave per row of 256) ----------------
__global__ __launch_bounds__(256)
void add_ln(const float* __restrict__ x, const float* __restrict__ d,
            float* __restrict__ y, const float* __restrict__ gs,
            const float* __restrict__ gb) {
  const int lane = threadIdx.x & 31, w = threadIdx.x >> 5;
  const size_t row = (size_t)blockIdx.x * 8 + w;
  const float* xr = x + row * E_;
  const float* dr = d + row * E_;
  float v[8]; float sum = 0.f;
#pragma unroll
  for (int i = 0; i < 8; ++i) { int c = lane + 32 * i; v[i] = xr[c] + dr[c]; sum += v[i]; }
#pragma unroll
  for (int off = 1; off < 32; off <<= 1) sum += __shfl_xor(sum, off, 32);
  float mean = sum * (1.f / E_);
  float var = 0.f;
#pragma unroll
  for (int i = 0; i < 8; ++i) { float u = v[i] - mean; var += u * u; }
#pragma unroll
  for (int off = 1; off < 32; off <<= 1) var += __shfl_xor(var, off, 32);
  float rs = rsqrtf(var * (1.f / E_) + 1e-5f);
  float* yr = y + row * E_;
#pragma unroll
  for (int i = 0; i < 8; ++i) {
    int c = lane + 32 * i;
    yr[c] = (v[i] - mean) * rs * gs[c] + gb[c];
  }
}

// ---------------- embedding gather (float4) ----------------
__global__ __launch_bounds__(256)
void embed_kernel(const int* __restrict__ tok, const float* __restrict__ emb,
                  float* __restrict__ x) {
  size_t idx4 = (size_t)blockIdx.x * 256 + threadIdx.x;   // over B*S*(E/4)
  size_t row = idx4 >> 6; int c4 = (int)(idx4 & 63) << 2;
  int tk = tok[row];
  *(float4*)(x + row * E_ + c4) = *(const float4*)(emb + (size_t)tk * E_ + c4);
}

// ---------------- per-sample sigmoid weight + tf weighting ----------------
__global__ void comp_weight(const float* __restrict__ feat, const float* __restrict__ fwW,
                            const float* __restrict__ fwb, const float* __restrict__ tf,
                            const int* __restrict__ lens, const int* __restrict__ lens_user,
                            float* __restrict__ weight) {
  int b = blockIdx.x, s = threadIdx.x;
  float d = fwb[0];
#pragma unroll
  for (int i = 0; i < 13; ++i) d += feat[b * 13 + i] * fwW[i];
  float wu = 1.f / (1.f + __expf(-d));
  float side = (s < lens_user[b]) ? wu : 1.f - wu;
  float valid = (s < lens[b]) ? 1.f : 0.f;
  weight[b * S_ + s] = tf[b * S_ + s] * side * valid;
}

// ---------------- weighted pooling + user/item gather -> output [B,768] ----------------
__global__ __launch_bounds__(256)
void build_output(const float* __restrict__ x, const float* __restrict__ weight,
                  const float* __restrict__ uemb, const float* __restrict__ iemb,
                  const int* __restrict__ uid, const int* __restrict__ iid,
                  float* __restrict__ outvec) {
  int b = blockIdx.x, e = threadIdx.x;
  const float* xb = x + (size_t)b * S_ * E_;
  float acc = 0.f;
  for (int s = 0; s < S_; ++s) acc += weight[b * S_ + s] * xb[(size_t)s * E_ + e];
  outvec[(size_t)b * O3E_ + E_ + e]     = acc;
  outvec[(size_t)b * O3E_ + e]          = uemb[(size_t)uid[b] * E_ + e];
  outvec[(size_t)b * O3E_ + 2 * E_ + e] = iemb[(size_t)iid[b] * E_ + e];
}

// ---------------- logits: one wave per (b, target) dot of length 768 ----------------
__global__ __launch_bounds__(256)
void logits_kernel(const float* __restrict__ oemb, const float* __restrict__ outvec,
                   const int* __restrict__ pos_t, const int* __restrict__ neg_t,
                   float* __restrict__ dst) {
  int lane = threadIdx.x & 31, w = threadIdx.x >> 5;
  int gid = blockIdx.x * 8 + w;                 // 0 .. B*T-1
  int b = gid / T_, j = gid % T_;
  int tgt = (j < P_) ? pos_t[b * P_ + j] : neg_t[b * NN_ + (j - P_)];
  const float* er = oemb + (size_t)tgt * O3E_;
  const float* ov = outvec + (size_t)b * O3E_;
  float acc = 0.f;
#pragma unroll
  for (int i = 0; i < O3E_ / 32; ++i) { int c = lane + 32 * i; acc += er[c] * ov[c]; }
#pragma unroll
  for (int off = 1; off < 32; off <<= 1) acc += __shfl_xor(acc, off, 32);
  if (lane == 0) dst[(size_t)b * T_ + j] = acc;
}

// ---------------- mask + new_targets ----------------
__global__ void mask_kernel(const int* __restrict__ pos_lens, const int* __restrict__ neg_lens,
                            float* __restrict__ dst) {
  int idx = blockIdx.x * 256 + threadIdx.x;     // over B*T
  int b = idx / T_, j = idx % T_;
  bool m = (j < P_) ? (j < pos_lens[b]) : ((j - P_) < neg_lens[b]);
  dst[(size_t)B_ * T_ + idx]     = m ? 1.f : 0.f;
  dst[(size_t)2 * B_ * T_ + idx] = (j < P_ && m) ? 1.f : 0.f;
}

extern "C" void kernel_launch(void* const* d_in, const int* in_sizes, int n_in,
                              void* d_out, int out_size, void* d_ws, size_t ws_size,
                              hipStream_t stream) {
  (void)in_sizes; (void)n_in; (void)out_size; (void)ws_size;
  const int*   attr        = (const int*)  d_in[0];
  const float* attr_tf     = (const float*)d_in[2];
  const float* attr_feat   = (const float*)d_in[3];
  const int*   attr_lens   = (const int*)  d_in[4];
  const int*   lens_user   = (const int*)  d_in[5];
  const int*   user_ids    = (const int*)  d_in[7];
  const int*   item_ids    = (const int*)  d_in[8];
  const int*   pos_targets = (const int*)  d_in[9];
  const int*   pos_lens    = (const int*)  d_in[10];
  const int*   neg_targets = (const int*)  d_in[11];
  const int*   neg_lens    = (const int*)  d_in[12];
  const float* attr_emb    = (const float*)d_in[13];
  const float* user_emb    = (const float*)d_in[14];
  const float* item_emb    = (const float*)d_in[15];
  const float* out_emb     = (const float*)d_in[16];
  const float* fw_W        = (const float*)d_in[17];
  const float* fw_b        = (const float*)d_in[18];
  const float* qkv_w       = (const float*)d_in[19];
  const float* qkv_b       = (const float*)d_in[20];
  const float* attn_out_w  = (const float*)d_in[21];
  const float* attn_out_b  = (const float*)d_in[22];
  const float* ln1_s       = (const float*)d_in[23];
  const float* ln1_b       = (const float*)d_in[24];
  const float* ff1_w       = (const float*)d_in[25];
  const float* ff1_b       = (const float*)d_in[26];
  const float* ff2_w       = (const float*)d_in[27];
  const float* ff2_b       = (const float*)d_in[28];
  const float* ln2_s       = (const float*)d_in[29];
  const float* ln2_b       = (const float*)d_in[30];
  float* out = (float*)d_out;

  const int M = B_ * S_;                       // 32768
  // workspace layout (floats)
  float* WX   = (float*)d_ws;                  // [M,E]        33.5 MB
  float* WB1  = WX  + (size_t)M * E_;          // [M,F] max    134 MB   (qkv / proj / ffn-hidden)
  float* WB2  = WB1 + (size_t)M * F_;          // [M,E]        33.5 MB  (attn-out / ffn-out)
  float* WW   = WB2 + (size_t)M * E_;          // [B,S] weights
  float* WOUT = WW  + (size_t)B_ * S_;         // [B,768] final feature

  // 1) embedding gather
  embed_kernel<<<(M * (E_ / 4)) / 256, 256, 0, stream>>>(attr, attr_emb, WX);

  // 2) transformer layers
  for (int l = 0; l < L_; ++l) {
    const float* qw  = qkv_w      + (size_t)l * 3 * E_ * E_;
    const float* qb  = qkv_b      + (size_t)l * 3 * E_;
    const float* aw  = attn_out_w + (size_t)l * E_ * E_;
    const float* ab  = attn_out_b + (size_t)l * E_;
    const float* l1s = ln1_s + (size_t)l * E_;
    const float* l1b = ln1_b + (size_t)l * E_;
    const float* f1w = ff1_w + (size_t)l * F_ * E_;
    const float* f1b = ff1_b + (size_t)l * F_;
    const float* f2w = ff2_w + (size_t)l * E_ * F_;
    const float* f2b = ff2_b + (size_t)l * E_;
    const float* l2s = ln2_s + (size_t)l * E_;
    const float* l2b = ln2_b + (size_t)l * E_;

    // qkv = X @ qkv_w^T + b   [M,768]
    gemm_bt<false><<<dim3(M / 128, (3 * E_) / 64), 256, 0, stream>>>(WX, qw, qb, WB1, M, 3 * E_, E_);
    // attention -> WB2 [M,E]
    attention_kernel<<<B_ * H_, 256, 0, stream>>>(WB1, attr_lens, WB2);
    // proj = attn @ W^T + b -> WB1 [M,E]
    gemm_bt<false><<<dim3(M / 128, E_ / 64), 256, 0, stream>>>(WB2, aw, ab, WB1, M, E_, E_);
    // X = LN(X + proj)
    add_ln<<<M / 8, 256, 0, stream>>>(WX, WB1, WX, l1s, l1b);
    // h = relu(X @ ff1^T + b) -> WB1 [M,F]
    gemm_bt<true><<<dim3(M / 128, F_ / 64), 256, 0, stream>>>(WX, f1w, f1b, WB1, M, F_, E_);
    // f2 = h @ ff2^T + b -> WB2 [M,E]
    gemm_bt<false><<<dim3(M / 128, E_ / 64), 256, 0, stream>>>(WB1, f2w, f2b, WB2, M, E_, F_);
    // X = LN(X + f2)
    add_ln<<<M / 8, 256, 0, stream>>>(WX, WB2, WX, l2s, l2b);
  }

  // 3) pooling weights, output vector, logits, mask/targets
  comp_weight<<<B_, S_, 0, stream>>>(attr_feat, fw_W, fw_b, attr_tf, attr_lens, lens_user, WW);
  build_output<<<B_, E_, 0, stream>>>(WX, WW, user_emb, item_emb, user_ids, item_ids, WOUT);
  logits_kernel<<<(B_ * T_) / 8, 256, 0, stream>>>(out_emb, WOUT, pos_targets, neg_targets, out);
  mask_kernel<<<(B_ * T_) / 256, 256, 0, stream>>>(pos_lens, neg_lens, out);
}
